// DeepForest_30356828848186
// MI455X (gfx1250) — matrix operand order repaired and checked by reading, not compiled, not run on hardware
//
#include <hip/hip_runtime.h>
#include <hip/hip_bf16.h>
#include <stdint.h>

// Problem constants (from reference setup_inputs)
#define T_   20
#define M_   31     // gated nodes per tree
#define D_   5      // depth
#define LEAF 16     // leaves per tree = 2^(D-1)
#define F_   256    // features
#define N_   128    // features per node
#define H_   64     // hidden
#define NC   10     // classes

#define PADH 8              // pad halfs per 256-half column (16B) -> 4-bank lane stride
#define COLW (F_ + PADH)    // 264 halfs per column in LDS

typedef __attribute__((ext_vector_type(16))) _Float16 v16h;
typedef __attribute__((ext_vector_type(8)))  _Float16 v8h;
typedef __attribute__((ext_vector_type(8)))  float    v8f;

// wave-relative LDS byte address of a __shared__ object
#define LDS_ADDR(p) ((uint32_t)(uintptr_t)(__attribute__((address_space(3))) void*)(void*)(p))

// ---------------- prep kernels ----------------

__global__ void cvt_x_f16(const float* __restrict__ x, _Float16* __restrict__ xh, int n) {
    for (int i = blockIdx.x * blockDim.x + threadIdx.x; i < n; i += gridDim.x * blockDim.x)
        xh[i] = (_Float16)x[i];
}

__global__ void zero_h16(_Float16* __restrict__ p, int n) {
    for (int i = blockIdx.x * blockDim.x + threadIdx.x; i < n; i += gridDim.x * blockDim.x)
        p[i] = (_Float16)0.0f;
}

__global__ void zero_i32(int* __restrict__ p, int n) {
    for (int i = blockIdx.x * blockDim.x + threadIdx.x; i < n; i += gridDim.x * blockDim.x)
        p[i] = 0;
}

// scatter W1 [T,M,n,H] (f32) into dense per-node column-major fp16 W1f [T*M][H][F]
// W1f[tm][j][f] = W1[tm][i][j] where f = feats[tm][i]
__global__ void scatter_w1(const float* __restrict__ W1, const int* __restrict__ feats,
                           _Float16* __restrict__ w1f) {
    int gid = blockIdx.x * blockDim.x + threadIdx.x;
    int total = T_ * M_ * N_;
    if (gid >= total) return;
    int i  = gid % N_;
    int tm = gid / N_;
    int f  = feats[(size_t)tm * N_ + i];
    const float* src = W1 + ((size_t)tm * N_ + i) * H_;
    _Float16* dst = w1f + (size_t)tm * H_ * F_ + f;
#pragma unroll
    for (int j = 0; j < H_; ++j)
        dst[(size_t)j * F_] = (_Float16)src[j];
}

// ---------------- async staging: one 32KB node tile -> padded LDS buffer ----------------
// Block (128 threads) covers 2048 16B-chunks in 16 rounds; chunk c = col*32 + (k/8):
// LDS half-offset = col*COLW + (c%32)*8  (column-padded layout).
__device__ __forceinline__ void stage_node(const _Float16* __restrict__ gsrc,
                                           uint32_t lds_base, int tid) {
#pragma unroll
    for (int r = 0; r < 16; ++r) {
        int c    = r * 128 + tid;
        int col  = c >> 5;
        int koff = (c & 31) << 3;
        uint64_t gaddr = (uint64_t)(uintptr_t)gsrc + (uint64_t)c * 16u;
        uint32_t laddr = lds_base + (uint32_t)(col * COLW + koff) * 2u;
        asm volatile("global_load_async_to_lds_b128 %0, %1, off"
                     :: "v"(laddr), "v"(gaddr) : "memory");
    }
}

// ---------------- main WMMA kernel ----------------
// grid = T * (B/64) blocks, 128 threads (4 waves); each wave owns 16 samples.
// All waves share the LDS-staged W1f node tile (4x less global traffic).
__global__ __launch_bounds__(128) void forest_main(
    const _Float16* __restrict__ xh,    // [B][F] fp16
    const _Float16* __restrict__ w1f,   // [T*M][H][F] fp16, zero-expanded
    const float* __restrict__ b1,       // [T*M][H]
    const float* __restrict__ w2,       // [T*M][H]
    const float* __restrict__ b2,       // [T*M]
    const int* __restrict__ bleft,      // [T][LEAF]
    const int* __restrict__ bright,     // [T][LEAF]
    int* __restrict__ counts,           // [B][NC]
    int tilesPerTree)
{
    __shared__ _Float16 wbuf[2][H_ * COLW];   // 2 x 33792 B double buffer
    __shared__ float gl[4][M_][16];           // gate logits per wave

    const int tid  = threadIdx.x;
    const int bx   = blockIdx.x;
    const int t    = bx / tilesPerTree;
    const int tile = bx % tilesPerTree;
    const int wave = tid >> 5;
    const int lane = tid & 31;
    const int half = lane >> 4;         // 0: lanes 0-15, 1: lanes 16-31
    const int l15  = lane & 15;
    const int bbase = tile * 64 + wave * 16;

    const uint32_t lbuf0 = LDS_ADDR(&wbuf[0][0]);
    const uint32_t lbuf1 = LDS_ADDR(&wbuf[1][0]);

    // ---- load A fragments (x tile: 16 samples x K=256) once, reused for all 31 nodes ----
    // A 16x32 layout (matches B): lanes 0-15 row=l15, K=kb..kb+15;
    //                            lanes 16-31 row=l15, K=kb+16..kb+31.
    const _Float16* xrow = xh + (size_t)(bbase + l15) * F_;
    const int lo = half ? 16 : 0;
    v16h afrag[8];
#pragma unroll
    for (int kc = 0; kc < 8; ++kc) {
        const _Float16* p = xrow + kc * 32 + lo;
        v8h a0 = *(const v8h*)(p);
        v8h a1 = *(const v8h*)(p + 8);
        v16h a;
#pragma unroll
        for (int e = 0; e < 8; ++e) { a[e] = a0[e]; a[e + 8] = a1[e]; }
        afrag[kc] = a;
    }

    // prime the pipeline: stage node 0 into buffer 0
    stage_node(w1f + (size_t)(t * M_ + 0) * H_ * F_, lbuf0, tid);

    // ---- per-node gate logits via WMMA (double-buffered LDS staging) ----
    for (int m = 0; m < M_; ++m) {
        const int tm = t * M_ + m;
        if (m + 1 < M_) {
            stage_node(w1f + (size_t)(tm + 1) * H_ * F_,
                       ((m + 1) & 1) ? lbuf1 : lbuf0, tid);
            // 16 newly issued + 16 for current tile; async loads complete in order,
            // so <=16 outstanding means the current tile has landed.
            asm volatile("s_wait_asynccnt 0x10" ::: "memory");
        } else {
            asm volatile("s_wait_asynccnt 0x0" ::: "memory");
        }
        __syncthreads();   // tile visible to all 4 waves

        const _Float16* wnode = &wbuf[m & 1][0];
        const float* b1n = b1 + (size_t)tm * H_;
        const float* w2n = w2 + (size_t)tm * H_;

        float part[8];
#pragma unroll
        for (int v = 0; v < 8; ++v) part[v] = 0.0f;

#pragma unroll
        for (int jt = 0; jt < 4; ++jt) {
            const int j = jt * 16 + l15;
            // B 32x16 layout used here (contiguous k, same as A): lanes 0-15 col=l15, K=kb..kb+15 ; lanes 16-31 col, K=kb+16..kb+31
            const _Float16* wcol = wnode + (size_t)j * COLW + half * 16;
            v8f acc = {};
#pragma unroll
            for (int kc = 0; kc < 8; ++kc) {
                v16h bfrag = *(const v16h*)(wcol + kc * 32);   // ds_load_b128 x2, conflict-free
                acc = __builtin_amdgcn_wmma_f32_16x16x32_f16(
                    /*neg_a=*/false, afrag[kc], /*neg_b=*/false, bfrag,
                    /*c_mod=*/(short)0, acc, /*reuse_a=*/false, /*reuse_b=*/false);
            }
            // fuse: h = relu(acc + b1[j]); partial += h * w2[j]
            const float b1j = b1n[j];
            const float w2j = w2n[j];
#pragma unroll
            for (int v = 0; v < 8; ++v) {
                float h = acc[v] + b1j;  // D: lanes<16 -> (M=v, N=l15); lanes>=16 -> (M=v+8, N=l15)
                h = h > 0.0f ? h : 0.0f;
                part[v] += h * w2j;
            }
        }
        // reduce over the 16 columns held within each 16-lane half
#pragma unroll
        for (int v = 0; v < 8; ++v) {
            float s = part[v];
            s += __shfl_xor(s, 1);
            s += __shfl_xor(s, 2);
            s += __shfl_xor(s, 4);
            s += __shfl_xor(s, 8);
            part[v] = s;
        }
        if (l15 == 0) {
            const float b2v = b2[tm];
#pragma unroll
            for (int v = 0; v < 8; ++v)
                gl[wave][m][half * 8 + v] = part[v] + b2v;   // logit; gate>0.5 <=> logit>0
        }
        __syncthreads();   // all reads of wbuf[m&1] done before it is restaged
    }

    // ---- heap traversal + vote (lanes 0-15 of each wave: one sample each) ----
    if (lane < 16) {
        const int s = lane;
        int idx = 1;
#pragma unroll
        for (int d = 0; d < D_ - 1; ++d) {
            float g = gl[wave][idx - 1][s];
            idx = (g > 0.0f) ? (2 * idx) : (2 * idx + 1);
        }
        float gleaf = gl[wave][idx - 1][s];
        int leaf = idx - LEAF;
        int pred = (gleaf > 0.0f) ? bleft[t * LEAF + leaf] : bright[t * LEAF + leaf];
        atomicAdd(counts + (size_t)(bbase + s) * NC + pred, 1);
    }
}

// ---------------- vote argmax ----------------
__global__ void vote_argmax(const int* __restrict__ counts, float* __restrict__ out, int B) {
    int b = blockIdx.x * blockDim.x + threadIdx.x;
    if (b >= B) return;
    const int* c = counts + (size_t)b * NC;
    int best = 0, bc = c[0];
#pragma unroll
    for (int k = 1; k < NC; ++k) {
        int v = c[k];
        if (v > bc) { bc = v; best = k; }   // strict > : ties -> smallest label
    }
    out[b] = (float)best;
}

// ---------------- launch ----------------
extern "C" void kernel_launch(void* const* d_in, const int* in_sizes, int n_in,
                              void* d_out, int out_size, void* d_ws, size_t ws_size,
                              hipStream_t stream) {
    const float* x     = (const float*)d_in[0];   // [B,F]
    const float* W1    = (const float*)d_in[1];   // [T,M,n,H]
    const float* b1    = (const float*)d_in[2];   // [T,M,H]
    const float* W2    = (const float*)d_in[3];   // [T,M,H]
    const float* b2    = (const float*)d_in[4];   // [T,M]
    const int*   feats = (const int*)d_in[5];     // [T,M,n]
    const int*   bl    = (const int*)d_in[6];     // [T,LEAF]
    const int*   br    = (const int*)d_in[7];     // [T,LEAF]
    float* out = (float*)d_out;

    const int B = in_sizes[0] / F_;               // 8192

    // workspace layout (256B-aligned)
    const size_t xh_bytes   = (size_t)B * F_ * sizeof(_Float16);           // 4 MB
    const size_t w1f_bytes  = (size_t)T_ * M_ * H_ * F_ * sizeof(_Float16);// ~19.4 MB
    char* ws = (char*)d_ws;
    _Float16* xh  = (_Float16*)ws;
    _Float16* w1f = (_Float16*)(ws + ((xh_bytes + 255) & ~(size_t)255));
    int* counts   = (int*)((char*)w1f + ((w1f_bytes + 255) & ~(size_t)255));

    const int nx   = B * F_;
    const int nw1f = T_ * M_ * H_ * F_;
    const int ncnt = B * NC;

    cvt_x_f16<<<(nx + 255) / 256, 256, 0, stream>>>(x, xh, nx);
    zero_h16<<<1024, 256, 0, stream>>>(w1f, nw1f);
    zero_i32<<<(ncnt + 255) / 256, 256, 0, stream>>>(counts, ncnt);
    scatter_w1<<<(T_ * M_ * N_ + 127) / 128, 128, 0, stream>>>(W1, feats, w1f);

    const int tilesPerTree = B / 64;
    forest_main<<<T_ * tilesPerTree, 128, 0, stream>>>(
        xh, w1f, b1, W2, b2, bl, br, counts, tilesPerTree);

    vote_argmax<<<(B + 255) / 256, 256, 0, stream>>>(counts, out, B);
}
